// MultiModalFeatureDenoiser_39092792328333
// MI455X (gfx1250) — matrix operand order repaired and checked by reading, not compiled, not run on hardware
//
#include <hip/hip_runtime.h>
#include <hip/hip_bf16.h>
#include <math.h>

typedef unsigned short u16;
typedef unsigned int   u32;
typedef __attribute__((ext_vector_type(16))) __bf16 bf16x16;
typedef __attribute__((ext_vector_type(8)))  float  f32x8;
typedef __attribute__((ext_vector_type(8)))  u32    u32x8;
typedef __attribute__((__vector_size__(16))) int i32x4v;

#define LAYERS 4
#define DMODEL 512
#define SEQ    1024
#define NB     8
#define NH     8
#define DH     64
#define FF     2048
#define NTOK   (NB*SEQ)   /* 8192 tokens */

#if defined(__has_builtin)
# if __has_builtin(__builtin_amdgcn_global_load_async_to_lds_b128)
#  define HAVE_ASYNC_LDS 1
# endif
# if __has_builtin(__builtin_amdgcn_s_wait_asynccnt)
#  define HAVE_WAIT_ASYNC 1
# endif
#endif

// ---------- helpers ----------
__device__ __forceinline__ u16 f2bf(float f){
  u32 u = __builtin_bit_cast(u32, f);
  u32 r = u + 0x7fffu + ((u >> 16) & 1u);   // RNE
  return (u16)(r >> 16);
}
__device__ __forceinline__ bf16x16 frag8(const u32* d){
  u32x8 v;
#pragma unroll
  for (int i = 0; i < 8; i++) v[i] = d[i];
  return __builtin_bit_cast(bf16x16, v);
}
__device__ __forceinline__ u32 ihash(u32 x){
  x ^= x >> 16; x *= 0x7feb352du; x ^= x >> 15; x *= 0x846ca68bu; x ^= x >> 16;
  return x;
}

// VALU-only cross-lane reductions (wave32): DPP quad_perm for xor1/2,
// v_permlane16 for xor4/8 (within 16-lane rows), v_permlanex16 for xor16.
__device__ __forceinline__ float lx1(float x){
  return __builtin_bit_cast(float, __builtin_amdgcn_update_dpp(0, __builtin_bit_cast(int, x), 0xB1, 0xF, 0xF, true));
}
__device__ __forceinline__ float lx2(float x){
  return __builtin_bit_cast(float, __builtin_amdgcn_update_dpp(0, __builtin_bit_cast(int, x), 0x4E, 0xF, 0xF, true));
}
__device__ __forceinline__ float lx4(float x){
  int i = __builtin_bit_cast(int, x);
  return __builtin_bit_cast(float, __builtin_amdgcn_permlane16(i, i, 0x32107654u, 0xBA98FEDCu, true, false));
}
__device__ __forceinline__ float lx8(float x){
  int i = __builtin_bit_cast(int, x);
  return __builtin_bit_cast(float, __builtin_amdgcn_permlane16(i, i, 0xFEDCBA98u, 0x76543210u, true, false));
}
__device__ __forceinline__ float lx16(float x){
  int i = __builtin_bit_cast(int, x);
  return __builtin_bit_cast(float, __builtin_amdgcn_permlanex16(i, i, 0x76543210u, 0xFEDCBA98u, true, false));
}
__device__ __forceinline__ float red_max16(float x){
  x = fmaxf(x, lx1(x)); x = fmaxf(x, lx2(x));
  x = fmaxf(x, lx4(x)); x = fmaxf(x, lx8(x));
  return x;
}
__device__ __forceinline__ float red_sum16(float x){
  x += lx1(x); x += lx2(x); x += lx4(x); x += lx8(x);
  return x;
}
__device__ __forceinline__ float red_sum32(float x){
  x = red_sum16(x);
  return x + lx16(x);
}

// ---------- async global -> LDS staging (CDNA5 ASYNCcnt path) ----------
__device__ __forceinline__ void ldg_to_lds_b128(const void* g, void* l){
#if defined(HAVE_ASYNC_LDS)
  __builtin_amdgcn_global_load_async_to_lds_b128(
      (__attribute__((address_space(1))) i32x4v*)(unsigned long long)g,
      (__attribute__((address_space(3))) i32x4v*)(unsigned int)(unsigned long long)l,
      0, 0);
#else
  *(uint4*)l = *(const uint4*)g;
#endif
}
__device__ __forceinline__ void wait_async_all(){
#if defined(HAVE_ASYNC_LDS)
# if defined(HAVE_WAIT_ASYNC)
  __builtin_amdgcn_s_wait_asynccnt(0);
# else
  asm volatile("s_wait_asynccnt 0" ::: "memory");
# endif
#endif
}

// ---------- f32 -> bf16 convert ----------
__global__ void cvt_bf16(const float* __restrict__ in, u16* __restrict__ out, int n){
  int i = blockIdx.x * blockDim.x + threadIdx.x;
  if (i < n) out[i] = f2bf(in[i]);
}

// ---------- DDIM closed-form noising ----------
__global__ void ddim_noise_init(const float* __restrict__ f, float* __restrict__ xF,
                                u16* __restrict__ xB, float cA, float sigma, int n){
  int i = blockIdx.x * blockDim.x + threadIdx.x;
  if (i >= n) return;
  u32 a = ihash((u32)i * 2654435761u + 0x9e3779b9u);
  u32 b = ihash((u32)i * 2246822519u + 0x85ebca6bu);
  float u1 = (float)((a >> 8) + 1u) * (1.0f/16777216.0f);
  float u2 = (float)(b >> 8) * (1.0f/16777216.0f);
  float r = sqrtf(fmaxf(-2.0f * __logf(u1), 0.0f));
  float z = r * __cosf(6.28318530718f * u2);
  float val = cA * f[i] + sigma * z;
  xF[i] = val; xB[i] = f2bf(val);
}

// ---------- tiled bf16 WMMA GEMM: C[M,N] = X[M,K] * W[N,K]^T + bias ----------
// block = 128 threads (4 waves, 2x2), block tile 128x128, wave tile 64x64.
// Double-buffered LDS: async-load stage k+1 while WMMAs run on stage k.
// EPI: 0 = f32 out, 1 = bf16 out, 2 = bf16 out + ReLU
template<int EPI>
__global__ __launch_bounds__(128) void gemm_x_wT(
    const u16* __restrict__ X, const u16* __restrict__ W,
    const float* __restrict__ bias, float* __restrict__ outF,
    u16* __restrict__ outB, int M, int N, int K)
{
  __shared__ u16 sA[2][128 * 72];   // 64 K-halves + 8 pad (144B row stride, 16B aligned)
  __shared__ u16 sB[2][128 * 72];

  const int tid  = threadIdx.x;
  const int wave = tid >> 5, lane = tid & 31;
  const int lr = lane & 15, hi = lane >> 4;
  const int wm = wave >> 1, wn = wave & 1;
  const int m0 = blockIdx.y * 128, n0 = blockIdx.x * 128;

  auto stage = [&](int k0, int buf){
#pragma unroll
    for (int i = 0; i < 8; i++){
      int idx = tid + i * 128;              // 1024 16B chunks per tile
      int row = idx >> 3, ch = idx & 7;
      ldg_to_lds_b128(X + (size_t)(m0 + row) * K + k0 + ch * 8, &sA[buf][row * 72 + ch * 8]);
      ldg_to_lds_b128(W + (size_t)(n0 + row) * K + k0 + ch * 8, &sB[buf][row * 72 + ch * 8]);
    }
  };

  f32x8 acc[4][4];
#pragma unroll
  for (int a = 0; a < 4; a++)
#pragma unroll
    for (int b = 0; b < 4; b++)
#pragma unroll
      for (int j = 0; j < 8; j++) acc[a][b][j] = 0.0f;

  stage(0, 0);
  int cur = 0;
  for (int k0 = 0; k0 < K; k0 += 64){
    wait_async_all();            // this wave's stage-k batch landed
    __syncthreads();             // all waves done with compute(k-1) and staging(k)
    if (k0 + 64 < K) stage(k0 + 64, cur ^ 1);   // overlap next loads with WMMAs

#pragma unroll
    for (int c = 0; c < 2; c++){
      bf16x16 af[4], bfr[4];
#pragma unroll
      for (int ms = 0; ms < 4; ms++){        // A-fragment gather: k = 16*hi + s (contiguous, as B)
        const u32* ar = (const u32*)&sA[cur][(wm * 64 + ms * 16 + lr) * 72];
        u32 d[8];
#pragma unroll
        for (int v = 0; v < 4; v++){
          d[v]     = ar[c * 16 + v + hi * 8];
          d[4 + v] = ar[c * 16 + 4 + v + hi * 8];
        }
        af[ms] = frag8(d);
      }
#pragma unroll
      for (int ns = 0; ns < 4; ns++){        // B-fragment: 32 contiguous bytes/lane
        const u16* br = &sB[cur][(wn * 64 + ns * 16 + lr) * 72 + c * 32 + hi * 16];
        uint4 lo = *(const uint4*)br;
        uint4 h4 = *(const uint4*)(br + 8);
        u32 d[8] = { lo.x, lo.y, lo.z, lo.w, h4.x, h4.y, h4.z, h4.w };
        bfr[ns] = frag8(d);
      }
#pragma unroll
      for (int ms = 0; ms < 4; ms++)
#pragma unroll
        for (int ns = 0; ns < 4; ns++)
          acc[ms][ns] = __builtin_amdgcn_wmma_f32_16x16x32_bf16(
              false, af[ms], false, bfr[ns], (short)0, acc[ms][ns], false, false);
    }
    cur ^= 1;
  }

#pragma unroll
  for (int ns = 0; ns < 4; ns++){
    const int col = n0 + wn * 64 + ns * 16 + lr;
    const float bv = bias ? bias[col] : 0.0f;
#pragma unroll
    for (int ms = 0; ms < 4; ms++){
#pragma unroll
      for (int j = 0; j < 8; j++){
        const int row = m0 + wm * 64 + ms * 16 + j + hi * 8;
        float v = acc[ms][ns][j] + bv;
        if (EPI == 2) v = fmaxf(v, 0.0f);
        if (EPI == 0) outF[(size_t)row * N + col] = v;
        else          outB[(size_t)row * N + col] = f2bf(v);
      }
    }
  }
}

// ---------- flash attention: per (b,h), 16 queries/wave, 32 keys/iter ----------
// Double-buffered K/V tiles; K via async LDS path, V transposed on ds store.
__global__ __launch_bounds__(128) void flash_attn(
    const u16* __restrict__ Q, const u16* __restrict__ Km,
    const u16* __restrict__ Vm, u16* __restrict__ O)
{
  __shared__ u16  sK[2][32 * 72];      // [key][dh], 144B row stride
  __shared__ u16  sVt[2][64 * 40];     // [dh][key] (transposed), 80B row stride
  __shared__ float sP[4][16 * 34];     // per-wave P staging, 136B row stride

  const int tid  = threadIdx.x;
  const int wave = tid >> 5, lane = tid & 31;
  const int lr = lane & 15, hi = lane >> 4;
  const int bh = blockIdx.y;
  const int b = bh >> 3, h = bh & 7;
  const int q0 = blockIdx.x * 64 + wave * 16;
  const size_t rowbase = (size_t)b * SEQ;
  const float scale = 0.125f;   // 1/sqrt(64)

  auto stage = [&](int kt, int buf){
#pragma unroll
    for (int i = 0; i < 2; i++){   // K tile: 32x64 bf16 via async LDS path
      int idx = tid + i * 128;
      int row = idx >> 3, ch = idx & 7;
      ldg_to_lds_b128(Km + (rowbase + kt + row) * DMODEL + h * DH + ch * 8,
                      &sK[buf][row * 72 + ch * 8]);
    }
#pragma unroll
    for (int i = 0; i < 8; i++){   // V tile transposed on store: sVt[dh][key]
      int w = tid + i * 128;
      int key = w >> 5, dp = w & 31;
      u32 u = *(const u32*)(Vm + (rowbase + kt + key) * DMODEL + h * DH + dp * 2);
      sVt[buf][(dp * 2    ) * 40 + key] = (u16)(u & 0xffffu);
      sVt[buf][(dp * 2 + 1) * 40 + key] = (u16)(u >> 16);
    }
  };

  bf16x16 qf[2];                 // Q rows as A-fragments over dh (2 chunks of 32)
  {
    const u32* qr = (const u32*)(Q + (rowbase + q0 + lr) * DMODEL + h * DH);
#pragma unroll
    for (int c = 0; c < 2; c++){
      u32 d[8];
#pragma unroll
      for (int v = 0; v < 4; v++){ d[v] = qr[c*16 + v + hi*8]; d[4+v] = qr[c*16 + 4 + v + hi*8]; }
      qf[c] = frag8(d);
    }
  }

  f32x8 o_[4];
  float m_r[8], l_r[8];
#pragma unroll
  for (int ns = 0; ns < 4; ns++)
#pragma unroll
    for (int j = 0; j < 8; j++) o_[ns][j] = 0.0f;
#pragma unroll
  for (int j = 0; j < 8; j++){ m_r[j] = -1e30f; l_r[j] = 0.0f; }

  float* Pw = sP[wave];

  stage(0, 0);
  int cur = 0;
  for (int kt = 0; kt < SEQ; kt += 32){
    wait_async_all();
    __syncthreads();
    if (kt + 32 < SEQ) stage(kt + 32, cur ^ 1);   // overlap next tile with compute

    // scores: Q(16xdh) * K^T -> two 16x16 tiles, contraction over dh=64
    f32x8 s0, s1;
#pragma unroll
    for (int j = 0; j < 8; j++){ s0[j] = 0.0f; s1[j] = 0.0f; }
#pragma unroll
    for (int nsub = 0; nsub < 2; nsub++){
#pragma unroll
      for (int c = 0; c < 2; c++){
        const u16* kr = &sK[cur][(nsub * 16 + lr) * 72 + c * 32 + hi * 16];
        uint4 lo = *(const uint4*)kr;
        uint4 h4 = *(const uint4*)(kr + 8);
        u32 d[8] = { lo.x, lo.y, lo.z, lo.w, h4.x, h4.y, h4.z, h4.w };
        bf16x16 kf = frag8(d);
        if (nsub == 0) s0 = __builtin_amdgcn_wmma_f32_16x16x32_bf16(false, qf[c], false, kf, (short)0, s0, false, false);
        else           s1 = __builtin_amdgcn_wmma_f32_16x16x32_bf16(false, qf[c], false, kf, (short)0, s1, false, false);
      }
    }

    // online softmax: VALU-only row reductions (DPP + permlane16)
    float fac[8];
#pragma unroll
    for (int j = 0; j < 8; j++){
      float a = s0[j] * scale, c = s1[j] * scale;
      float mx = red_max16(fmaxf(a, c));
      float mnew = fmaxf(m_r[j], mx);
      fac[j] = __expf(m_r[j] - mnew);
      m_r[j] = mnew;
      float p0 = __expf(a - mnew), p1 = __expf(c - mnew);
      s0[j] = p0; s1[j] = p1;
      float rs = red_sum16(p0 + p1);
      l_r[j] = l_r[j] * fac[j] + rs;
#pragma unroll
      for (int ns = 0; ns < 4; ns++) o_[ns][j] *= fac[j];
    }

    // D-layout -> A-layout via per-wave LDS bounce, f32 -> bf16
#pragma unroll
    for (int j = 0; j < 8; j++){
      Pw[(j + hi * 8) * 34 + lr]      = s0[j];
      Pw[(j + hi * 8) * 34 + 16 + lr] = s1[j];
    }
    asm volatile("s_wait_dscnt 0" ::: "memory");
    bf16x16 pf;
    {
      const float* pr = Pw + lr * 34;
      u32 d[8];
#pragma unroll
      for (int v = 0; v < 4; v++){
        float2 c0 = *(const float2*)(pr + 2 * v + hi * 16);
        float2 c1 = *(const float2*)(pr + 8 + 2 * v + hi * 16);
        d[v]     = (u32)f2bf(c0.x) | ((u32)f2bf(c0.y) << 16);
        d[4 + v] = (u32)f2bf(c1.x) | ((u32)f2bf(c1.y) << 16);
      }
      pf = frag8(d);
    }

    // O += P(16x32) * V(32x64): 4 WMMAs over dh columns
#pragma unroll
    for (int ns = 0; ns < 4; ns++){
      const u16* vr = &sVt[cur][(ns * 16 + lr) * 40 + hi * 16];
      uint4 lo = *(const uint4*)vr;
      uint4 h4 = *(const uint4*)(vr + 8);
      u32 d[8] = { lo.x, lo.y, lo.z, lo.w, h4.x, h4.y, h4.z, h4.w };
      bf16x16 vf = frag8(d);
      o_[ns] = __builtin_amdgcn_wmma_f32_16x16x32_bf16(false, pf, false, vf, (short)0, o_[ns], false, false);
    }
    cur ^= 1;
  }

#pragma unroll
  for (int ns = 0; ns < 4; ns++)
#pragma unroll
    for (int j = 0; j < 8; j++){
      float v = o_[ns][j] / l_r[j];
      O[(rowbase + q0 + j + hi * 8) * DMODEL + h * DH + ns * 16 + lr] = f2bf(v);
    }
}

// ---------- fused residual + layernorm (1 wave per row) ----------
__global__ __launch_bounds__(256) void add_layernorm(
    const float* __restrict__ x, const float* __restrict__ o,
    const float* __restrict__ g, const float* __restrict__ bta,
    float* __restrict__ xoutF, u16* __restrict__ xoutB)
{
  const int wave = threadIdx.x >> 5, lane = threadIdx.x & 31;
  const int row = blockIdx.x * 8 + wave;
  const size_t base = (size_t)row * DMODEL;
  float v[16];
  float s = 0.0f, ss = 0.0f;
#pragma unroll
  for (int i = 0; i < 16; i++){
    int c = lane + i * 32;
    float t = x[base + c] + o[base + c];
    v[i] = t; s += t; ss += t * t;
  }
  s = red_sum32(s); ss = red_sum32(ss);
  float mean = s * (1.0f / DMODEL);
  float var  = ss * (1.0f / DMODEL) - mean * mean;
  float rstd = rsqrtf(var + 1e-5f);
#pragma unroll
  for (int i = 0; i < 16; i++){
    int c = lane + i * 32;
    float y = (v[i] - mean) * rstd * g[c] + bta[c];
    xoutF[base + c] = y;
    xoutB[base + c] = f2bf(y);
  }
}

// ---------- host ----------
extern "C" void kernel_launch(void* const* d_in, const int* in_sizes, int n_in,
                              void* d_out, int out_size, void* d_ws, size_t ws_size,
                              hipStream_t stream)
{
  (void)in_sizes; (void)n_in; (void)out_size; (void)ws_size;
  const float* features = (const float*)d_in[0];
  const float* Wq   = (const float*)d_in[1];
  const float* Wk   = (const float*)d_in[2];
  const float* Wv   = (const float*)d_in[3];
  const float* bq   = (const float*)d_in[4];
  const float* bk   = (const float*)d_in[5];
  const float* bv   = (const float*)d_in[6];
  const float* Wo   = (const float*)d_in[7];
  const float* bo   = (const float*)d_in[8];
  const float* ln1g = (const float*)d_in[9];
  const float* ln1b = (const float*)d_in[10];
  const float* ln2g = (const float*)d_in[11];
  const float* ln2b = (const float*)d_in[12];
  const float* W1   = (const float*)d_in[13];
  const float* b1   = (const float*)d_in[14];
  const float* W2   = (const float*)d_in[15];
  const float* b2   = (const float*)d_in[16];
  const float* Wout = (const float*)d_in[17];
  const float* bout = (const float*)d_in[18];

  char* ws = (char*)d_ws; size_t off = 0;
  auto alloc = [&](size_t bytes) -> void* {
    void* p = ws + off; off += (bytes + 255) & ~(size_t)255; return p;
  };
  float* x_f32 = (float*)alloc((size_t)NTOK * DMODEL * 4);
  u16*   x_bf  = (u16*)  alloc((size_t)NTOK * DMODEL * 2);
  u16*   q_bf  = (u16*)  alloc((size_t)NTOK * DMODEL * 2);
  u16*   k_bf  = (u16*)  alloc((size_t)NTOK * DMODEL * 2);
  u16*   v_bf  = (u16*)  alloc((size_t)NTOK * DMODEL * 2);
  u16*   a_bf  = (u16*)  alloc((size_t)NTOK * DMODEL * 2);
  float* tmp_f = (float*)alloc((size_t)NTOK * DMODEL * 4);
  u16*   hid_bf= (u16*)  alloc((size_t)NTOK * FF * 2);
  u16*   wq_bf = (u16*)  alloc((size_t)LAYERS * DMODEL * DMODEL * 2);
  u16*   wk_bf = (u16*)  alloc((size_t)LAYERS * DMODEL * DMODEL * 2);
  u16*   wv_bf = (u16*)  alloc((size_t)LAYERS * DMODEL * DMODEL * 2);
  u16*   wo_bf = (u16*)  alloc((size_t)LAYERS * DMODEL * DMODEL * 2);
  u16*   w1_bf = (u16*)  alloc((size_t)LAYERS * FF * DMODEL * 2);
  u16*   w2_bf = (u16*)  alloc((size_t)LAYERS * DMODEL * FF * 2);
  u16*   wout_bf=(u16*)  alloc((size_t)DMODEL * DMODEL * 2);

  // DDIM closed-form coefficients (host, double precision)
  double ac[1000]; double prod = 1.0;
  for (int t = 0; t < 1000; t++){
    double beta = 1e-4 + (0.02 - 1e-4) * ((double)t / 999.0);
    prod *= (1.0 - beta); ac[t] = prod;
  }
  double var = 0.0, logA = 0.0;
  for (int t = 1; t < 1000; t++){ var = ac[t] * var + (1.0 - ac[t]); logA += 0.5 * log(ac[t]); }
  float cA = (float)exp(logA);        // underflows to ~0
  float sigma = (float)sqrt(var);

  auto cvt = [&](const float* src, u16* dst, size_t n){
    cvt_bf16<<<dim3((unsigned)((n + 255) / 256)), 256, 0, stream>>>(src, dst, (int)n);
  };
  cvt(Wq, wq_bf, (size_t)LAYERS * DMODEL * DMODEL);
  cvt(Wk, wk_bf, (size_t)LAYERS * DMODEL * DMODEL);
  cvt(Wv, wv_bf, (size_t)LAYERS * DMODEL * DMODEL);
  cvt(Wo, wo_bf, (size_t)LAYERS * DMODEL * DMODEL);
  cvt(W1, w1_bf, (size_t)LAYERS * FF * DMODEL);
  cvt(W2, w2_bf, (size_t)LAYERS * DMODEL * FF);
  cvt(Wout, wout_bf, (size_t)DMODEL * DMODEL);

  const int n = NTOK * DMODEL;
  ddim_noise_init<<<dim3((n + 255) / 256), 256, 0, stream>>>(features, x_f32, x_bf, cA, sigma, n);

  const dim3 gD(DMODEL / 128, NTOK / 128);   // (4, 64)
  for (int l = 0; l < LAYERS; l++){
    const size_t wOff = (size_t)l * DMODEL * DMODEL;
    gemm_x_wT<1><<<gD, 128, 0, stream>>>(x_bf, wq_bf + wOff, bq + l * DMODEL, nullptr, q_bf, NTOK, DMODEL, DMODEL);
    gemm_x_wT<1><<<gD, 128, 0, stream>>>(x_bf, wk_bf + wOff, bk + l * DMODEL, nullptr, k_bf, NTOK, DMODEL, DMODEL);
    gemm_x_wT<1><<<gD, 128, 0, stream>>>(x_bf, wv_bf + wOff, bv + l * DMODEL, nullptr, v_bf, NTOK, DMODEL, DMODEL);

    flash_attn<<<dim3(SEQ / 64, NB * NH), 128, 0, stream>>>(q_bf, k_bf, v_bf, a_bf);

    gemm_x_wT<0><<<gD, 128, 0, stream>>>(a_bf, wo_bf + wOff, bo + l * DMODEL, tmp_f, nullptr, NTOK, DMODEL, DMODEL);
    add_layernorm<<<dim3(NTOK / 8), 256, 0, stream>>>(x_f32, tmp_f, ln1g + l * DMODEL, ln1b + l * DMODEL, x_f32, x_bf);

    gemm_x_wT<2><<<dim3(FF / 128, NTOK / 128), 128, 0, stream>>>(x_bf, w1_bf + (size_t)l * FF * DMODEL, b1 + l * FF, nullptr, hid_bf, NTOK, FF, DMODEL);
    gemm_x_wT<0><<<gD, 128, 0, stream>>>(hid_bf, w2_bf + (size_t)l * DMODEL * FF, b2 + l * DMODEL, tmp_f, nullptr, NTOK, DMODEL, FF);
    add_layernorm<<<dim3(NTOK / 8), 256, 0, stream>>>(x_f32, tmp_f, ln2g + l * DMODEL, ln2b + l * DMODEL, x_f32, x_bf);
  }

  gemm_x_wT<0><<<gD, 128, 0, stream>>>(x_bf, wout_bf, bout, (float*)d_out, nullptr, NTOK, DMODEL, DMODEL);
}
